// GSABlock_46901042872806
// MI455X (gfx1250) — compile-verified, matrix operand order checked by reading, not run on hardware
//
#include <hip/hip_runtime.h>
#include <hip/hip_bf16.h>

// GSA block for MI455X (gfx1250, wave32, WMMA).
// B=4, T=2048, D=512, H=4, K=V=128, M=64.

#define BB 4
#define TT 2048
#define DD 512
#define HH 4
#define KK 128
#define VV 128
#define MM 64
#define BT (BB * TT)          // 8192 rows for all GEMMs

#define LDA 40                // padded LDS row length (bf16 elems): 80B, 16B-aligned runs

typedef __attribute__((ext_vector_type(16))) __bf16 v16bf;
typedef __attribute__((ext_vector_type(8)))  __bf16 v8bf;
typedef __attribute__((ext_vector_type(2)))  __bf16 v2bf;
typedef __attribute__((ext_vector_type(8)))  float  v8f;

#if __has_builtin(__builtin_amdgcn_cvt_pk_bf16_f32)
__device__ __forceinline__ unsigned pack2bf(float a, float b) {
    v2bf r = __builtin_amdgcn_cvt_pk_bf16_f32(a, b);   // v_cvt_pk_bf16_f32
    return __builtin_bit_cast(unsigned, r);
}
#else
__device__ __forceinline__ unsigned pack2bf(float a, float b) {
    unsigned ua = __builtin_bit_cast(unsigned, a);
    unsigned ub = __builtin_bit_cast(unsigned, b);
    ua = (ua + 0x7FFFu + ((ua >> 16) & 1u)) >> 16;     // RNE
    ub = (ub + 0x7FFFu + ((ub >> 16) & 1u)) >> 16;
    return ua | (ub << 16);
}
#endif

__device__ __forceinline__ __bf16 f2bf(float f) {
    union { unsigned u; __bf16 h[2]; } x;
    x.u = pack2bf(f, 0.f);
    return x.h[0];
}

// ---------------------------------------------------------------------------
// Tiled GEMM: Y(8192 x N) = act( X(8192 x 512) @ W(512 x N) ), bf16 WMMA.
// Block = 256 threads (8 waves). Block tile = 128(M) x 64(N), K-step = 32.
// Software-pipelined double-buffered LDS: fetch tile k+1 to registers, WMMA
// tile k from LDS[cur], then stage registers into LDS[cur^1]; ONE barrier
// per K-step. A staged row-major As[row][k]; B staged transposed Bst[col][k]
// so every lane fragment is two contiguous 16B runs -> ds_load_b128.
// ACT: 0 = none, 1 = swish, 2 = ge = exp(log_sigmoid(x)/8)
// ---------------------------------------------------------------------------
template <int ACT>
__global__ __launch_bounds__(256) void gemm_wmma_kernel(
    const float* __restrict__ X, const float* __restrict__ W,
    float* __restrict__ Y, int N) {
    __shared__ __align__(16) __bf16 As[2][128 * LDA];   // 2 x 10.0 KB
    __shared__ __align__(16) __bf16 Bst[2][64 * LDA];   // 2 x  5.0 KB

    const int tid  = threadIdx.x;
    const int lane = tid & 31;
    const int wv   = tid >> 5;
    const int rowBase = blockIdx.y * 128;
    const int colBase = blockIdx.x * 64;

    const int mrow = lane & 15;            // row of 16-row tile / col of 16-col tile
    const int kb   = (lane >> 4) << 3;     // 0 or 8: K sub-block for this half-wave

    const float4* X4 = (const float4*)X;
    const float4* W4 = (const float4*)W;

    // Per-thread staging coordinates (constant)
    int arr[4], ac[4];                     // A: row, float4-col
#pragma unroll
    for (int i = 0; i < 4; ++i) { int e = tid + i * 256; arr[i] = e >> 3; ac[i] = e & 7; }
    int brr[2], bc[2];                     // B: k-row, float4-col
#pragma unroll
    for (int i = 0; i < 2; ++i) { int e = tid + i * 256; brr[i] = e >> 4; bc[i] = e & 15; }

    float4 ra[4], rb[2];

    // ---- prologue: fetch tile 0 and stage into buffer 0
#pragma unroll
    for (int i = 0; i < 4; ++i)
        ra[i] = X4[(((rowBase + arr[i]) * DD) >> 2) + ac[i]];
#pragma unroll
    for (int i = 0; i < 2; ++i)
        rb[i] = W4[((brr[i] * N + colBase) >> 2) + bc[i]];
#pragma unroll
    for (int i = 0; i < 4; ++i) {
        uint2 t; t.x = pack2bf(ra[i].x, ra[i].y); t.y = pack2bf(ra[i].z, ra[i].w);
        *(uint2*)&As[0][arr[i] * LDA + ac[i] * 4] = t;
    }
#pragma unroll
    for (int i = 0; i < 2; ++i) {
        int c = bc[i] * 4, r = brr[i];
        Bst[0][(c + 0) * LDA + r] = f2bf(rb[i].x);
        Bst[0][(c + 1) * LDA + r] = f2bf(rb[i].y);
        Bst[0][(c + 2) * LDA + r] = f2bf(rb[i].z);
        Bst[0][(c + 3) * LDA + r] = f2bf(rb[i].w);
    }

    v8f acc[4] = {};
    int cur = 0;

    for (int k0 = 0; k0 < DD; k0 += 32) {
        __syncthreads();                      // LDS[cur] ready; LDS[cur^1] free
        const bool more = (k0 + 32 < DD);
        if (more) {
            // Issue next-tile global loads early: they fly under the WMMAs.
#pragma unroll
            for (int i = 0; i < 4; ++i)
                ra[i] = X4[(((rowBase + arr[i]) * DD + k0 + 32) >> 2) + ac[i]];
#pragma unroll
            for (int i = 0; i < 2; ++i)
                rb[i] = W4[(((k0 + 32 + brr[i]) * N + colBase) >> 2) + bc[i]];
            if (k0 + 64 < DD)                 // gfx1250 global_prefetch_b8
                __builtin_prefetch(&X[(rowBase + (tid >> 1)) * DD + k0 + 64], 0, 1);
        }

        // ---- WMMA on buffer `cur`
        const __bf16* Ab = &As[cur][0];
        const __bf16* Bb = &Bst[cur][0];
        const int arow = (wv * 16 + mrow) * LDA;
        v8bf alo = *(const v8bf*)&Ab[arow + kb];
        v8bf ahi = *(const v8bf*)&Ab[arow + 16 + kb];
        v16bf a = __builtin_shufflevector(alo, ahi,
            0, 1, 2, 3, 4, 5, 6, 7, 8, 9, 10, 11, 12, 13, 14, 15);
#pragma unroll
        for (int jn = 0; jn < 4; ++jn) {
            const int bcol = (jn * 16 + mrow) * LDA;
            v8bf blo = *(const v8bf*)&Bb[bcol + kb];
            v8bf bhi = *(const v8bf*)&Bb[bcol + 16 + kb];
            v16bf bfr = __builtin_shufflevector(blo, bhi,
                0, 1, 2, 3, 4, 5, 6, 7, 8, 9, 10, 11, 12, 13, 14, 15);
            acc[jn] = __builtin_amdgcn_wmma_f32_16x16x32_bf16(
                false, a, false, bfr, (short)0, acc[jn], false, false);
        }

        if (more) {
            // ---- stage fetched registers into buffer `cur^1`
            __bf16* An = &As[cur ^ 1][0];
            __bf16* Bn = &Bst[cur ^ 1][0];
#pragma unroll
            for (int i = 0; i < 4; ++i) {
                uint2 t; t.x = pack2bf(ra[i].x, ra[i].y); t.y = pack2bf(ra[i].z, ra[i].w);
                *(uint2*)&An[arr[i] * LDA + ac[i] * 4] = t;
            }
#pragma unroll
            for (int i = 0; i < 2; ++i) {
                int c = bc[i] * 4, r = brr[i];
                Bn[(c + 0) * LDA + r] = f2bf(rb[i].x);
                Bn[(c + 1) * LDA + r] = f2bf(rb[i].y);
                Bn[(c + 2) * LDA + r] = f2bf(rb[i].z);
                Bn[(c + 3) * LDA + r] = f2bf(rb[i].w);
            }
        }
        cur ^= 1;
    }

    // Epilogue: C/D layout -> (row = base + r + 8*(lane>=16), col = base + 16*jn + lane%16)
    const int mhalf = (lane >> 4) << 3;
#pragma unroll
    for (int jn = 0; jn < 4; ++jn) {
#pragma unroll
        for (int r = 0; r < 8; ++r) {
            int row = rowBase + wv * 16 + mhalf + r;
            int col = colBase + jn * 16 + mrow;
            float x = acc[jn][r];
            if (ACT == 1) {
                x = x / (1.f + __expf(-x));                      // swish
            } else if (ACT == 2) {
                float ax = fabsf(x);
                float ls = fminf(x, 0.f) - log1pf(__expf(-ax));  // log_sigmoid, stable
                x = __expf(ls * 0.125f);                         // ge = exp(g/8)
            }
            Y[row * N + col] = x;
        }
    }
}

// ---------------------------------------------------------------------------
// Pass 1 scan: hk[k,m] = hk*ge[m] + k[k]*(1-ge[m]); ok[m] = sum_k q[k]*hk[k,m]
// One block per (b,h). 256 threads: thread t owns m = t&63, k-group t>>6 (32 k's).
// ---------------------------------------------------------------------------
__global__ __launch_bounds__(256) void scan1_kernel(
    const float* __restrict__ q, const float* __restrict__ k,
    const float* __restrict__ ge, float* __restrict__ ok) {
    const int b = blockIdx.x >> 2, h = blockIdx.x & 3;
    const int t = threadIdx.x;
    const int m = t & 63, kb = (t >> 6) * 32;

    __shared__ float qs[128], ks[128], red[256];
    float hk[32];
#pragma unroll
    for (int i = 0; i < 32; ++i) hk[i] = 0.f;

    for (int step = 0; step < TT; ++step) {
        int rowq = (b * TT + step) * (HH * KK) + h * KK;
        int rowg = (b * TT + step) * (HH * MM) + h * MM;
        if (t < 128) qs[t] = q[rowq + t];
        else         ks[t - 128] = k[rowq + (t - 128)];
        float gm = ge[rowg + m];
        __syncthreads();
        float sm = 1.f - gm;
        float acc = 0.f;
#pragma unroll
        for (int i = 0; i < 32; ++i) {
            hk[i] = hk[i] * gm + ks[kb + i] * sm;
            acc += qs[kb + i] * hk[i];
        }
        red[t] = acc;
        __syncthreads();
        if (t < 64)
            ok[rowg + t] = red[t] + red[64 + t] + red[128 + t] + red[192 + t];
        __syncthreads();
    }
}

// ---------------------------------------------------------------------------
// Softmax over M=64 slots; one wave per row (32768 rows), 8 rows per block.
// ---------------------------------------------------------------------------
__global__ __launch_bounds__(256) void softmax_kernel(
    const float* __restrict__ ok, float* __restrict__ p) {
    const int lane = threadIdx.x & 31, wv = threadIdx.x >> 5;
    const int row = blockIdx.x * 8 + wv;
    const float* src = ok + row * MM;
    float a = src[lane], b = src[lane + 32];
    float mx = fmaxf(a, b);
#pragma unroll
    for (int off = 16; off > 0; off >>= 1) mx = fmaxf(mx, __shfl_xor(mx, off, 32));
    float ea = __expf(a - mx), eb = __expf(b - mx);
    float s = ea + eb;
#pragma unroll
    for (int off = 16; off > 0; off >>= 1) s += __shfl_xor(s, off, 32);
    float inv = 1.f / s;
    p[row * MM + lane]      = ea * inv;
    p[row * MM + lane + 32] = eb * inv;
}

// ---------------------------------------------------------------------------
// Pass 2 scan: hv[m,v] = hv*ge[m] + (1-ge[m])*v[v]; o[v] = sum_m p[m]*hv[m,v]
// One block per (b,h). Thread t owns v-col t&127, m-group t>>7 (32 m's).
// ---------------------------------------------------------------------------
__global__ __launch_bounds__(256) void scan2_kernel(
    const float* __restrict__ p, const float* __restrict__ v,
    const float* __restrict__ ge, float* __restrict__ o) {
    const int b = blockIdx.x >> 2, h = blockIdx.x & 3;
    const int t = threadIdx.x;
    const int vc = t & 127, mb = (t >> 7) * 32;

    __shared__ float ps[64], ges[64], vs[128], red[256];
    float hv[32];
#pragma unroll
    for (int i = 0; i < 32; ++i) hv[i] = 0.f;

    for (int step = 0; step < TT; ++step) {
        int rowv = (b * TT + step) * (HH * VV) + h * VV;
        int rowg = (b * TT + step) * (HH * MM) + h * MM;
        if (t < 64)       ps[t] = p[rowg + t];
        else if (t < 128) ges[t - 64] = ge[rowg + (t - 64)];
        else              vs[t - 128] = v[rowv + (t - 128)];
        __syncthreads();
        float vv = vs[vc];
        float acc = 0.f;
#pragma unroll
        for (int i = 0; i < 32; ++i) {
            float g = ges[mb + i];
            hv[i] = hv[i] * g + (1.f - g) * vv;
            acc += ps[mb + i] * hv[i];
        }
        red[t] = acc;
        __syncthreads();
        if (t < 128) o[rowv + t] = red[t] + red[128 + t];
        __syncthreads();
    }
}

// ---------------------------------------------------------------------------
// Per-head RMSNorm over V=128; one wave per head-row (32768 rows), in place.
// ---------------------------------------------------------------------------
__global__ __launch_bounds__(256) void rmsnorm_kernel(
    float* __restrict__ o, const float* __restrict__ w) {
    const int lane = threadIdx.x & 31, wv = threadIdx.x >> 5;
    const int row = blockIdx.x * 8 + wv;
    float* base = o + row * VV;
    float x0 = base[lane], x1 = base[lane + 32], x2 = base[lane + 64], x3 = base[lane + 96];
    float ss = x0 * x0 + x1 * x1 + x2 * x2 + x3 * x3;
#pragma unroll
    for (int off = 16; off > 0; off >>= 1) ss += __shfl_xor(ss, off, 32);
    float r = rsqrtf(ss * (1.f / (float)VV) + 1e-5f);
    base[lane]      = x0 * r * w[lane];
    base[lane + 32] = x1 * r * w[lane + 32];
    base[lane + 64] = x2 * r * w[lane + 64];
    base[lane + 96] = x3 * r * w[lane + 96];
}

// ---------------------------------------------------------------------------

extern "C" void kernel_launch(void* const* d_in, const int* in_sizes, int n_in,
                              void* d_out, int out_size, void* d_ws, size_t ws_size,
                              hipStream_t stream) {
    const float* x      = (const float*)d_in[0];
    const float* Wq     = (const float*)d_in[1];
    const float* Wk     = (const float*)d_in[2];
    const float* Wv     = (const float*)d_in[3];
    const float* Wf     = (const float*)d_in[4];
    const float* gnormw = (const float*)d_in[5];
    const float* Wo     = (const float*)d_in[6];
    float* out = (float*)d_out;

    float* ws = (float*)d_ws;
    float* q  = ws;                       // BT*512
    float* k  = q  + (size_t)BT * 512;    // BT*512
    float* v  = k  + (size_t)BT * 512;    // BT*512
    float* ge = v  + (size_t)BT * 512;    // BT*256
    float* ok = ge + (size_t)BT * 256;    // BT*256
    float* p  = ok + (size_t)BT * 256;    // BT*256
    float* o  = p  + (size_t)BT * 256;    // BT*512

    dim3 blk(256);
    dim3 grd512(512 / 64, BT / 128);  // N=512 GEMMs
    dim3 grd256(256 / 64, BT / 128);  // N=256 GEMM (Wf)

    gemm_wmma_kernel<1><<<grd512, blk, 0, stream>>>(x, Wq, q, 512);   // swish
    gemm_wmma_kernel<1><<<grd512, blk, 0, stream>>>(x, Wk, k, 512);   // swish
    gemm_wmma_kernel<0><<<grd512, blk, 0, stream>>>(x, Wv, v, 512);   // none
    gemm_wmma_kernel<2><<<grd256, blk, 0, stream>>>(x, Wf, ge, 256);  // ge gate

    scan1_kernel<<<dim3(BB * HH), blk, 0, stream>>>(q, k, ge, ok);
    softmax_kernel<<<dim3((BT * HH) / 8), blk, 0, stream>>>(ok, p);
    scan2_kernel<<<dim3(BB * HH), blk, 0, stream>>>(p, v, ge, o);
    rmsnorm_kernel<<<dim3((BT * HH) / 8), blk, 0, stream>>>(o, gnormw);

    gemm_wmma_kernel<0><<<grd512, blk, 0, stream>>>(o, Wo, out, 512); // out proj
}
